// FluxIPAttnProcessor_72155450573119
// MI455X (gfx1250) — matrix operand order repaired and checked by reading, not compiled, not run on hardware
//
#include <hip/hip_runtime.h>

typedef unsigned short u16;
typedef unsigned int   u32;

typedef __attribute__((ext_vector_type(16))) __bf16 v16bf;
typedef __attribute__((ext_vector_type(8)))  float  v8f;

union Frag { v16bf v; u32 u[8]; };

__device__ inline float bf_lo(u32 u){ return __uint_as_float(u << 16); }
__device__ inline float bf_hi(u32 u){ return __uint_as_float(u & 0xffff0000u); }

// f32 -> bf16 using the hardware converter (RNE), not bit-twiddling.
__device__ inline u16 f2bf(float f){
  union { __bf16 h; u16 u; } x;
  x.h = (__bf16)f;
  return x.u;
}
__device__ inline u32 pack2(float a, float b){
#if __has_builtin(__builtin_amdgcn_cvt_pk_bf16_f32)
  auto r = __builtin_amdgcn_cvt_pk_bf16_f32(a, b);   // v_cvt_pk_bf16_f32
  return __builtin_bit_cast(u32, r);
#else
  return (u32)f2bf(a) | ((u32)f2bf(b) << 16);
#endif
}

__device__ inline v8f wmma_bf16(const Frag& a, const Frag& b, v8f c){
  return __builtin_amdgcn_wmma_f32_16x16x32_bf16(false, a.v, false, b.v, (short)0, c, false, false);
}

// ---------------------------------------------------------------------------
// Generic NT GEMM: Out[M,N] = A[M,K] * W[N,K]^T + bias[N]
// A: f32 or bf16 (template), W/bias: f32, Out: f32 or bf16 (template).
// Block: 256 threads (8 waves), tile 128x128, K-step 32.
// Wave grid 4x2; each wave owns 32x64 = 2x4 WMMA 16x16 tiles.
// Double-buffered LDS: next tile's global loads are issued before the WMMA
// clause, converted/stored to the alternate buffer after it (1 barrier/iter).
// ---------------------------------------------------------------------------
template<bool A_BF16, bool OUT_BF16>
__global__ __launch_bounds__(256) void gemm_kernel(
    const void* __restrict__ Ap, const float* __restrict__ W,
    const float* __restrict__ bias, void* __restrict__ Outp,
    int M, int N, int K)
{
  __shared__ u16 As[2][128][34];
  __shared__ u16 Bs[2][128][34];
  const int tid   = threadIdx.x;
  const int wave  = tid >> 5, lane = tid & 31;
  const int lane15 = lane & 15, hi = lane >> 4;
  const int waveM = wave >> 1, waveN = wave & 1;
  const int rowBase = blockIdx.y * 128;
  const int colBase = blockIdx.x * 128;

  int linR[4], linC[4];
  #pragma unroll
  for (int i=0;i<4;++i){ int lin = tid + i*256; linR[i] = lin >> 3; linC[i] = (lin & 7) << 2; }

  const v8f vzero = {0.f,0.f,0.f,0.f,0.f,0.f,0.f,0.f};
  v8f acc[2][4];
  #pragma unroll
  for (int i=0;i<2;++i)
    #pragma unroll
    for (int j=0;j<4;++j) acc[i][j] = vzero;

  float4 aF[4]; uint2 aB[4]; float4 wF[4];

  auto load_tiles = [&](int kb){
    if (A_BF16) {
      const u16* A = (const u16*)Ap;
      #pragma unroll
      for (int i=0;i<4;++i){
        int gr = rowBase + linR[i];
        uint2 z; z.x = 0u; z.y = 0u;
        aB[i] = z;
        if (gr < M) aB[i] = *(const uint2*)(A + (size_t)gr*K + kb + linC[i]);
      }
    } else {
      const float* A = (const float*)Ap;
      #pragma unroll
      for (int i=0;i<4;++i){
        int gr = rowBase + linR[i];
        float4 z = {0.f,0.f,0.f,0.f};
        aF[i] = z;
        if (gr < M) aF[i] = *(const float4*)(A + (size_t)gr*K + kb + linC[i]);
      }
    }
    #pragma unroll
    for (int i=0;i<4;++i){
      int gn = colBase + linR[i];
      float4 z = {0.f,0.f,0.f,0.f};
      wF[i] = z;
      if (gn < N) wF[i] = *(const float4*)(W + (size_t)gn*K + kb + linC[i]);
    }
  };
  auto store_tiles = [&](int buf){
    #pragma unroll
    for (int i=0;i<4;++i){
      if (A_BF16) {
        *(u32*)&As[buf][linR[i]][linC[i]]   = aB[i].x;
        *(u32*)&As[buf][linR[i]][linC[i]+2] = aB[i].y;
      } else {
        *(u32*)&As[buf][linR[i]][linC[i]]   = pack2(aF[i].x, aF[i].y);
        *(u32*)&As[buf][linR[i]][linC[i]+2] = pack2(aF[i].z, aF[i].w);
      }
      *(u32*)&Bs[buf][linR[i]][linC[i]]   = pack2(wF[i].x, wF[i].y);
      *(u32*)&Bs[buf][linR[i]][linC[i]+2] = pack2(wF[i].z, wF[i].w);
    }
  };

  // prologue: stage first tile
  load_tiles(0);
  store_tiles(0);
  __syncthreads();

  int cur = 0;
  for (int kb = 0; kb < K; kb += 32) {
    const bool more = (kb + 32) < K;
    if (more) load_tiles(kb + 32);   // global loads in flight during WMMA

    Frag af[2], bfr[4];
    #pragma unroll
    for (int mt=0;mt<2;++mt){
      int r = waveM*32 + mt*16 + lane15;
      #pragma unroll
      for (int i=0;i<8;++i){
        int p = ((i & 4) << 1) + hi*4 + (i & 3);   // A layout pair index
        af[mt].u[i] = *(const u32*)&As[cur][r][2*p];
      }
    }
    #pragma unroll
    for (int nt=0;nt<4;++nt){
      int n = waveN*64 + nt*16 + lane15;
      #pragma unroll
      for (int i=0;i<8;++i)
        bfr[nt].u[i] = *(const u32*)&Bs[cur][n][2*(((i & 4) << 1) + hi*4 + (i & 3))]; // B layout
    }
    #pragma unroll
    for (int mt=0;mt<2;++mt)
      #pragma unroll
      for (int nt=0;nt<4;++nt)
        acc[mt][nt] = wmma_bf16(af[mt], bfr[nt], acc[mt][nt]);

    if (more) {
      store_tiles(cur ^ 1);
      __syncthreads();
      cur ^= 1;
    }
  }

  // ---- epilogue: bias + store (C layout: reg j -> row j+8*hi, col lane15) ----
  #pragma unroll
  for (int mt=0;mt<2;++mt){
    #pragma unroll
    for (int nt=0;nt<4;++nt){
      int col = colBase + waveN*64 + nt*16 + lane15;
      float bv = bias[col];
      #pragma unroll
      for (int j=0;j<8;++j){
        int row = rowBase + waveM*32 + mt*16 + j + 8*hi;
        if (row < M){
          float val = acc[mt][nt][j] + bv;
          if (OUT_BF16) ((u16*)Outp)[(size_t)row*N + col] = f2bf(val);
          else          ((float*)Outp)[(size_t)row*N + col] = val;
        }
      }
    }
  }
}

// ---------------------------------------------------------------------------
// Prep kernels: RMSNorm (per 128-d head row) + RoPE + scale; bf16 in/out.
// One wave per (head, seq) row; 4 elements per lane (wave32 * 4 = 128).
// ---------------------------------------------------------------------------
__device__ inline float wave_sumsq(float v){
  float s = v;
  s += __shfl_xor(s, 16); s += __shfl_xor(s, 8);
  s += __shfl_xor(s, 4);  s += __shfl_xor(s, 2); s += __shfl_xor(s, 1);
  return s;
}

#define ATTN_SCALE 0.08838834764831845f

__global__ __launch_bounds__(256) void prep_main_kernel(
    const u16* __restrict__ q_lin, const u16* __restrict__ k_lin, const u16* __restrict__ v_lin,
    const u16* __restrict__ eq_lin, const u16* __restrict__ ek_lin, const u16* __restrict__ ev_lin,
    const float* __restrict__ rc, const float* __restrict__ rs,
    const float* __restrict__ nqw, const float* __restrict__ nkw,
    const float* __restrict__ naqw, const float* __restrict__ nakw,
    u16* __restrict__ Qm, u16* __restrict__ Km, u16* __restrict__ Vm)
{
  int wave = threadIdx.x >> 5, lane = threadIdx.x & 31;
  int r = blockIdx.x * 8 + wave;        // 0 .. 24*2560-1
  int h = r / 2560, s = r - h * 2560;
  int d0 = lane * 4;
  bool txt = s < 512;
  size_t srcoff = (txt ? (size_t)s : (size_t)(s - 512)) * 3072 + h*128 + d0;
  const u16* qs = (txt ? eq_lin : q_lin) + srcoff;
  const u16* ks = (txt ? ek_lin : k_lin) + srcoff;
  const u16* vs = (txt ? ev_lin : v_lin) + srcoff;
  const float* qw = (txt ? naqw : nqw) + d0;
  const float* kw = (txt ? nakw : nkw) + d0;
  const float* cp = rc + (size_t)s*128 + d0;
  const float* sp = rs + (size_t)s*128 + d0;
  float c0=cp[0], c1=cp[1], c2=cp[2], c3=cp[3];
  float s0=sp[0], s1=sp[1], s2=sp[2], s3=sp[3];
  size_t dst = ((size_t)h * 2560 + s) * 128 + d0;

  // --- Q: rms -> rope -> scale ---
  {
    uint2 raw = *(const uint2*)qs;
    float x0=bf_lo(raw.x), x1=bf_hi(raw.x), x2=bf_lo(raw.y), x3=bf_hi(raw.y);
    float ss = wave_sumsq(x0*x0 + x1*x1 + x2*x2 + x3*x3);
    float rr = rsqrtf(ss * (1.0f/128.0f) + 1e-6f);
    x0 *= rr*qw[0]; x1 *= rr*qw[1]; x2 *= rr*qw[2]; x3 *= rr*qw[3];
    float y0 = x0*c0 - x1*s0, y1 = x1*c1 + x0*s1;
    float y2 = x2*c2 - x3*s2, y3 = x3*c3 + x2*s3;
    uint2 o; o.x = pack2(y0*ATTN_SCALE, y1*ATTN_SCALE); o.y = pack2(y2*ATTN_SCALE, y3*ATTN_SCALE);
    *(uint2*)(Qm + dst) = o;
  }
  // --- K: rms -> rope ---
  {
    uint2 raw = *(const uint2*)ks;
    float x0=bf_lo(raw.x), x1=bf_hi(raw.x), x2=bf_lo(raw.y), x3=bf_hi(raw.y);
    float ss = wave_sumsq(x0*x0 + x1*x1 + x2*x2 + x3*x3);
    float rr = rsqrtf(ss * (1.0f/128.0f) + 1e-6f);
    x0 *= rr*kw[0]; x1 *= rr*kw[1]; x2 *= rr*kw[2]; x3 *= rr*kw[3];
    float y0 = x0*c0 - x1*s0, y1 = x1*c1 + x0*s1;
    float y2 = x2*c2 - x3*s2, y3 = x3*c3 + x2*s3;
    uint2 o; o.x = pack2(y0, y1); o.y = pack2(y2, y3);
    *(uint2*)(Km + dst) = o;
  }
  // --- V: copy ---
  *(uint2*)(Vm + dst) = *(const uint2*)vs;
}

__global__ __launch_bounds__(256) void prep_ipq_kernel(
    const u16* __restrict__ q_lin, const float* __restrict__ nipq,
    u16* __restrict__ Qip)
{
  int wave = threadIdx.x >> 5, lane = threadIdx.x & 31;
  int r = blockIdx.x * 8 + wave;        // 0 .. 24*2048-1
  int h = r >> 11, s = r & 2047;
  int d0 = lane * 4;
  const u16* src = q_lin + (size_t)s*3072 + h*128 + d0;
  uint2 raw = *(const uint2*)src;
  float x0=bf_lo(raw.x), x1=bf_hi(raw.x), x2=bf_lo(raw.y), x3=bf_hi(raw.y);
  float ss = wave_sumsq(x0*x0 + x1*x1 + x2*x2 + x3*x3);
  float rr = rsqrtf(ss * (1.0f/128.0f) + 1e-6f);
  const float* w = nipq + d0;
  uint2 o;
  o.x = pack2(x0*rr*w[0]*ATTN_SCALE, x1*rr*w[1]*ATTN_SCALE);
  o.y = pack2(x2*rr*w[2]*ATTN_SCALE, x3*rr*w[3]*ATTN_SCALE);
  *(uint2*)(Qip + ((size_t)h*2048 + s)*128 + d0) = o;
}

__global__ __launch_bounds__(256) void prep_ipkv_kernel(
    const u16* __restrict__ ipk_lin, const u16* __restrict__ ipv_lin,
    const float* __restrict__ nipk,
    u16* __restrict__ Kip, u16* __restrict__ Vip)
{
  int wave = threadIdx.x >> 5, lane = threadIdx.x & 31;
  int r = blockIdx.x * 8 + wave;        // 0 .. 24*64-1
  int h = r >> 6, s = r & 63;
  int d0 = lane * 4;
  size_t srcoff = (size_t)s*3072 + h*128 + d0;
  size_t dst = ((size_t)h*64 + s)*128 + d0;
  {
    uint2 raw = *(const uint2*)(ipk_lin + srcoff);
    float x0=bf_lo(raw.x), x1=bf_hi(raw.x), x2=bf_lo(raw.y), x3=bf_hi(raw.y);
    float ss = wave_sumsq(x0*x0 + x1*x1 + x2*x2 + x3*x3);
    float rr = rsqrtf(ss * (1.0f/128.0f) + 1e-6f);
    const float* w = nipk + d0;
    uint2 o; o.x = pack2(x0*rr*w[0], x1*rr*w[1]); o.y = pack2(x2*rr*w[2], x3*rr*w[3]);
    *(uint2*)(Kip + dst) = o;
  }
  *(uint2*)(Vip + dst) = *(const uint2*)(ipv_lin + srcoff);
}

// ---------------------------------------------------------------------------
// Flash attention (head-major bf16 Q/K/V, D=128). 128 queries / block,
// KV tiles of 64, online softmax. Scale pre-folded into Q.
// IS_MAIN: writes bf16 to attn_out[s, h*128+d], adding ip_out for s>=512.
// else   : writes f32 ip_out[s, h*128+d].
// ---------------------------------------------------------------------------
struct AttnLoopBuf {
  u16 Kt[64][136];     // K tile, row = kv, col = d
  u16 Vt[128][72];     // V tile transposed, row = d, col = kv
  u16 Ps[8][16][72];   // per-wave P (16 x 64), bf16
};
union __align__(16) AttnShared {
  u16 Qs[128][136];
  AttnLoopBuf lp;
};

template<bool IS_MAIN>
__global__ __launch_bounds__(256) void attn_kernel(
    const u16* __restrict__ Q, const u16* __restrict__ Kb, const u16* __restrict__ Vb,
    const float* __restrict__ ip_in,
    float* __restrict__ out_f32, u16* __restrict__ out_bf16,
    int q_len, int kv_len)
{
  __shared__ AttnShared sh;
  const int h = blockIdx.y;
  const int qbase = blockIdx.x * 128;
  const int tid = threadIdx.x;
  const int wave = tid >> 5, lane = tid & 31;
  const int lane15 = lane & 15, hi = lane >> 4;
  const u16* Qh = Q  + (size_t)h * q_len  * 128;
  const u16* Kh = Kb + (size_t)h * kv_len * 128;
  const u16* Vh = Vb + (size_t)h * kv_len * 128;

  // stage Q tile (128 x 128 bf16)
  #pragma unroll
  for (int i = 0; i < 8; ++i) {
    int lin = tid + i*256;                 // 0..2047 chunks of 8 bf16
    int r = lin >> 4, c8 = (lin & 15) * 8;
    *(uint4*)&sh.Qs[r][c8] = *(const uint4*)(Qh + (size_t)(qbase + r)*128 + c8);
  }
  __syncthreads();

  // extract Q fragments (wave owns rows wave*16 .. +15), keep in regs
  Frag qf[4];
  {
    int qrow = wave*16 + lane15;
    #pragma unroll
    for (int ks=0; ks<4; ++ks){
      int kb = ks*32;
      #pragma unroll
      for (int i=0;i<8;++i){
        int p = ((i & 4) << 1) + hi*4 + (i & 3);
        qf[ks].u[i] = *(const u32*)&sh.Qs[qrow][kb + 2*p];
      }
    }
  }
  __syncthreads();   // Qs dead; loop buffers may overwrite

  const v8f vzero = {0.f,0.f,0.f,0.f,0.f,0.f,0.f,0.f};
  v8f o[8];
  #pragma unroll
  for (int i=0;i<8;++i) o[i] = vzero;
  float mst[8], lst[8];
  #pragma unroll
  for (int j=0;j<8;++j){ mst[j] = -1e30f; lst[j] = 0.f; }

  for (int kv0 = 0; kv0 < kv_len; kv0 += 64) {
    // load K tile (64 x 128)
    #pragma unroll
    for (int i=0;i<4;++i){
      int lin = tid + i*256;               // 0..1023
      int r = lin >> 4, c8 = (lin & 15) * 8;
      *(uint4*)&sh.lp.Kt[r][c8] = *(const uint4*)(Kh + (size_t)(kv0 + r)*128 + c8);
    }
    // load V tile transposed (Vt[d][kv])
    #pragma unroll
    for (int i=0;i<4;++i){
      int lin = tid + i*256;
      int r = lin >> 4, c8 = (lin & 15) * 8;
      uint4 vv = *(const uint4*)(Vh + (size_t)(kv0 + r)*128 + c8);
      u32 wd[4] = {vv.x, vv.y, vv.z, vv.w};
      #pragma unroll
      for (int e=0;e<4;++e){
        sh.lp.Vt[c8 + 2*e    ][r] = (u16)(wd[e] & 0xffffu);
        sh.lp.Vt[c8 + 2*e + 1][r] = (u16)(wd[e] >> 16);
      }
    }
    __syncthreads();

    // S = Q @ K^T : wave computes 16 x 64
    v8f sacc[4];
    #pragma unroll
    for (int nt=0;nt<4;++nt) sacc[nt] = vzero;
    #pragma unroll
    for (int ks=0;ks<4;++ks){
      int kb = ks*32;
      #pragma unroll
      for (int nt=0;nt<4;++nt){
        Frag bfr;
        int n = nt*16 + lane15;
        #pragma unroll
        for (int i=0;i<8;++i)
          bfr.u[i] = *(const u32*)&sh.lp.Kt[n][kb + 2*(((i & 4) << 1) + hi*4 + (i & 3))];
        sacc[nt] = wmma_bf16(qf[ks], bfr, sacc[nt]);
      }
    }

    // online softmax (rows live in 16-lane halves of C layout)
    #pragma unroll
    for (int j=0;j<8;++j){
      float v0=sacc[0][j], v1=sacc[1][j], v2=sacc[2][j], v3=sacc[3][j];
      float mx = fmaxf(fmaxf(v0,v1), fmaxf(v2,v3));
      mx = fmaxf(mx, __shfl_xor(mx,1)); mx = fmaxf(mx, __shfl_xor(mx,2));
      mx = fmaxf(mx, __shfl_xor(mx,4)); mx = fmaxf(mx, __shfl_xor(mx,8));
      float mnew = fmaxf(mst[j], mx);
      float alpha = __expf(mst[j] - mnew);
      float p0=__expf(v0-mnew), p1=__expf(v1-mnew), p2=__expf(v2-mnew), p3=__expf(v3-mnew);
      float rsum = p0+p1+p2+p3;
      rsum += __shfl_xor(rsum,1); rsum += __shfl_xor(rsum,2);
      rsum += __shfl_xor(rsum,4); rsum += __shfl_xor(rsum,8);
      lst[j] = lst[j]*alpha + rsum;
      mst[j] = mnew;
      #pragma unroll
      for (int nt=0;nt<8;++nt) o[nt][j] *= alpha;
      int pr = j + 8*hi;
      sh.lp.Ps[wave][pr][ 0 + lane15] = f2bf(p0);
      sh.lp.Ps[wave][pr][16 + lane15] = f2bf(p1);
      sh.lp.Ps[wave][pr][32 + lane15] = f2bf(p2);
      sh.lp.Ps[wave][pr][48 + lane15] = f2bf(p3);
    }
    // Ps region is wave-private; same-wave DS ops complete in order.

    // O += P @ V : wave computes 16 x 128
    #pragma unroll
    for (int ks=0;ks<2;++ks){
      int kb = ks*32;
      Frag pa;
      #pragma unroll
      for (int i=0;i<8;++i){
        int p = ((i & 4) << 1) + hi*4 + (i & 3);
        pa.u[i] = *(const u32*)&sh.lp.Ps[wave][lane15][kb + 2*p];
      }
      #pragma unroll
      for (int nt=0;nt<8;++nt){
        Frag bfr;
        int n = nt*16 + lane15;
        #pragma unroll
        for (int i=0;i<8;++i)
          bfr.u[i] = *(const u32*)&sh.lp.Vt[n][kb + 2*(((i & 4) << 1) + hi*4 + (i & 3))];
        o[nt] = wmma_bf16(pa, bfr, o[nt]);
      }
    }
    __syncthreads();
  }

  // epilogue
  #pragma unroll
  for (int j=0;j<8;++j){
    float rinv = 1.0f / lst[j];
    int s = qbase + wave*16 + j + 8*hi;
    #pragma unroll
    for (int nt=0;nt<8;++nt){
      int d = nt*16 + lane15;
      float val = o[nt][j] * rinv;
      size_t col = (size_t)h*128 + d;
      if (IS_MAIN) {
        if (s >= 512) val += ip_in[(size_t)(s - 512)*3072 + col];
        out_bf16[(size_t)s*3072 + col] = f2bf(val);
      } else {
        out_f32[(size_t)s*3072 + col] = val;
      }
    }
  }
}

// ---------------------------------------------------------------------------
extern "C" void kernel_launch(void* const* d_in, const int* in_sizes, int n_in,
                              void* d_out, int out_size, void* d_ws, size_t ws_size,
                              hipStream_t stream)
{
  (void)in_sizes; (void)n_in; (void)out_size; (void)ws_size;
  const float* hs   = (const float*)d_in[0];
  const float* enc  = (const float*)d_in[1];
  const float* ip   = (const float*)d_in[2];
  const float* rc   = (const float*)d_in[3];
  const float* rs   = (const float*)d_in[4];
  const float* Wq   = (const float*)d_in[5];  const float* bq   = (const float*)d_in[6];
  const float* Wk   = (const float*)d_in[7];  const float* bk   = (const float*)d_in[8];
  const float* Wv   = (const float*)d_in[9];  const float* bv   = (const float*)d_in[10];
  const float* nqw  = (const float*)d_in[11]; const float* nkw  = (const float*)d_in[12];
  const float* Wqa  = (const float*)d_in[13]; const float* bqa  = (const float*)d_in[14];
  const float* Wka  = (const float*)d_in[15]; const float* bka  = (const float*)d_in[16];
  const float* Wva  = (const float*)d_in[17]; const float* bva  = (const float*)d_in[18];
  const float* naqw = (const float*)d_in[19]; const float* nakw = (const float*)d_in[20];
  const float* Wout = (const float*)d_in[21]; const float* bout = (const float*)d_in[22];
  const float* Wadd = (const float*)d_in[23]; const float* badd = (const float*)d_in[24];
  const float* Wkip = (const float*)d_in[25]; const float* bkip = (const float*)d_in[26];
  const float* Wvip = (const float*)d_in[27]; const float* bvip = (const float*)d_in[28];
  const float* nipq = (const float*)d_in[29]; const float* nipk = (const float*)d_in[30];

  char* wsp = (char*)d_ws;
  size_t off = 0;
  auto alloc = [&](size_t bytes)->void* {
    void* p = wsp + off;
    off = (off + bytes + 255) & ~((size_t)255);
    return p;
  };
  u16* q_lin   = (u16*)alloc((size_t)2048*3072*2);
  u16* k_lin   = (u16*)alloc((size_t)2048*3072*2);
  u16* v_lin   = (u16*)alloc((size_t)2048*3072*2);
  u16* eq_lin  = (u16*)alloc((size_t)512*3072*2);
  u16* ek_lin  = (u16*)alloc((size_t)512*3072*2);
  u16* ev_lin  = (u16*)alloc((size_t)512*3072*2);
  u16* ipk_lin = (u16*)alloc((size_t)64*3072*2);
  u16* ipv_lin = (u16*)alloc((size_t)64*3072*2);
  u16* Qm      = (u16*)alloc((size_t)24*2560*128*2);
  u16* Km      = (u16*)alloc((size_t)24*2560*128*2);
  u16* Vm      = (u16*)alloc((size_t)24*2560*128*2);
  u16* Qip     = (u16*)alloc((size_t)24*2048*128*2);
  u16* Kip     = (u16*)alloc((size_t)24*64*128*2);
  u16* Vip     = (u16*)alloc((size_t)24*64*128*2);
  float* ip_out  = (float*)alloc((size_t)2048*3072*4);
  u16*   attn_o  = (u16*)alloc((size_t)2560*3072*2);

  dim3 blk(256);

  // 1) projections (f32 in, bf16 out)
  gemm_kernel<false,true><<<dim3(24,16), blk, 0, stream>>>(hs,  Wq,  bq,  q_lin,  2048, 3072, 3072);
  gemm_kernel<false,true><<<dim3(24,16), blk, 0, stream>>>(hs,  Wk,  bk,  k_lin,  2048, 3072, 3072);
  gemm_kernel<false,true><<<dim3(24,16), blk, 0, stream>>>(hs,  Wv,  bv,  v_lin,  2048, 3072, 3072);
  gemm_kernel<false,true><<<dim3(24,4),  blk, 0, stream>>>(enc, Wqa, bqa, eq_lin, 512,  3072, 3072);
  gemm_kernel<false,true><<<dim3(24,4),  blk, 0, stream>>>(enc, Wka, bka, ek_lin, 512,  3072, 3072);
  gemm_kernel<false,true><<<dim3(24,4),  blk, 0, stream>>>(enc, Wva, bva, ev_lin, 512,  3072, 3072);
  gemm_kernel<false,true><<<dim3(24,1),  blk, 0, stream>>>(ip,  Wkip,bkip,ipk_lin,64,   3072, 4096);
  gemm_kernel<false,true><<<dim3(24,1),  blk, 0, stream>>>(ip,  Wvip,bvip,ipv_lin,64,   3072, 4096);

  // 2) RMSNorm + RoPE + scale -> head-major bf16 Q/K/V
  prep_main_kernel<<<dim3(7680), blk, 0, stream>>>(q_lin, k_lin, v_lin, eq_lin, ek_lin, ev_lin,
                                                   rc, rs, nqw, nkw, naqw, nakw, Qm, Km, Vm);
  prep_ipq_kernel<<<dim3(6144), blk, 0, stream>>>(q_lin, nipq, Qip);
  prep_ipkv_kernel<<<dim3(192), blk, 0, stream>>>(ipk_lin, ipv_lin, nipk, Kip, Vip);

  // 3) IP attention (2048 q x 64 kv) -> f32 ip_out
  attn_kernel<false><<<dim3(16,24), blk, 0, stream>>>(Qip, Kip, Vip, (const float*)nullptr,
                                                      ip_out, (u16*)nullptr, 2048, 64);
  // 4) main attention (2560 x 2560), fuses "+ ip_out" for image rows -> bf16 attn_o
  attn_kernel<true><<<dim3(20,24), blk, 0, stream>>>(Qm, Km, Vm, ip_out,
                                                     (float*)nullptr, attn_o, 2560, 2560);

  // 5) output projections (bf16 in, f32 out straight to d_out: img first, then enc)
  float* out_img = (float*)d_out;
  float* out_enc = out_img + (size_t)2048*3072;
  gemm_kernel<true,false><<<dim3(24,16), blk, 0, stream>>>(attn_o + (size_t)512*3072, Wout, bout,
                                                           out_img, 2048, 3072, 3072);
  gemm_kernel<true,false><<<dim3(24,4),  blk, 0, stream>>>(attn_o, Wadd, badd,
                                                           out_enc, 512, 3072, 3072);
}
